// SBGNN_68719476997
// MI455X (gfx1250) — matrix operand order repaired and checked by reading, compiled, not run on hardware
//
#include <hip/hip_runtime.h>
#include <hip/hip_bf16.h>

#define NA_ 100000
#define NB_ 100000
#define D_  32
#define E_  1000000
#define B_  8192
#define K_  10
#define REG_ 0.01f

typedef _Float16 v16h __attribute__((ext_vector_type(16)));
typedef _Float16 v8h  __attribute__((ext_vector_type(8)));
typedef float    v8f  __attribute__((ext_vector_type(8)));

__device__ __forceinline__ float wave_sum32(float v) {
#pragma unroll
  for (int m = 16; m > 0; m >>= 1) v += __shfl_xor(v, m, 32);
  return v;
}

__device__ __forceinline__ v16h cat8(v8h lo, v8h hi) {
  return __builtin_shufflevector(lo, hi, 0,1,2,3,4,5,6,7,8,9,10,11,12,13,14,15);
}

// 16x32 f16 A-operand (CDNA5 wave32 WMMA A layout) from row-major f16 matrix,
// row stride S halves, tile top row row0, K-chunk base kc.
__device__ __forceinline__ v16h load_a_tile(const _Float16* base, int row0, int S,
                                            int kc, int lane) {
  const _Float16* p = base + (size_t)(row0 + (lane & 15)) * S + kc + ((lane >> 4) << 3);
  v8h lo = *(const v8h*)p;
  v8h hi = *(const v8h*)(p + 16);
  return cat8(lo, hi);
}

// 32x16 f16 B-operand from a transposed weight copy Wt (Wt[j*S+k] = W[k][j]),
// row stride S halves, tile col base col0, K-chunk base kc.
__device__ __forceinline__ v16h load_b_tile(const _Float16* Wt, int col0, int S,
                                            int kc, int lane) {
  const _Float16* p = Wt + (size_t)(col0 + (lane & 15)) * S + kc + ((lane >> 4) << 3);
  v8h lo = *(const v8h*)p;
  v8h hi = *(const v8h*)(p + 16);
  return cat8(lo, hi);
}

__device__ __forceinline__ v8f wmma_f16(v16h a, v16h b, v8f c) {
  return __builtin_amdgcn_wmma_f32_16x16x32_f16(false, a, false, b, (short)0, c,
                                                false, false);
}

// ---------------- f32 -> f16 feature conversion ----------------
__global__ void cvt_f16_kernel(const float* __restrict__ in,
                               _Float16* __restrict__ out, int n) {
  int i = blockIdx.x * blockDim.x + threadIdx.x;
  if (i < n) out[i] = (_Float16)in[i];
}

// ---------------- new_emb = nbr @ W + b (WMMA) ----------------
__global__ void __launch_bounds__(256)
transform_kernel(const _Float16* __restrict__ nbr16, const float* __restrict__ W,
                 const float* __restrict__ bias, float* __restrict__ out, int nrows) {
  __shared__ __align__(16) _Float16 Wt[32 * 32];
  __shared__ float bs[32];
  int tid = threadIdx.x;
  for (int i = tid; i < 32 * 32; i += blockDim.x) {
    int k = i >> 5, j = i & 31;
    Wt[j * 32 + k] = (_Float16)W[i];       // W[k*32+j] -> Wt[j][k]
  }
  if (tid < 32) bs[tid] = bias[tid];
  __syncthreads();

  int lane   = tid & 31;
  int wave   = (blockIdx.x * blockDim.x + tid) >> 5;
  int nwaves = (gridDim.x * blockDim.x) >> 5;
  int ntiles = nrows >> 4;                  // 100000/16 = 6250, exact
  for (int t = wave; t < ntiles; t += nwaves) {
    int row0 = t << 4;
    v16h a = load_a_tile(nbr16, row0, 32, 0, lane);   // K = 32 in one shot
#pragma unroll
    for (int ct = 0; ct < 2; ++ct) {
      int col = ct * 16 + (lane & 15);
      v16h b = load_b_tile(Wt, ct * 16, 32, 0, lane);
      v8f c;
      float bv = bs[col];
#pragma unroll
      for (int j = 0; j < 8; ++j) c[j] = bv;
      c = wmma_f16(a, b, c);
      int rbase = row0 + ((lane >> 4) << 3);
#pragma unroll
      for (int j = 0; j < 8; ++j)
        out[(size_t)(rbase + j) * 32 + col] = c[j];
    }
  }
}

// ---------------- edge attention + segment-sum (wave per edge) ----------------
__global__ void __launch_bounds__(256)
edge_agg_kernel(const int* __restrict__ edges, const float* __restrict__ srcfeat,
                const float* __restrict__ nemb, const float* __restrict__ att,
                float* __restrict__ msg, float* __restrict__ rowsum, int ne) {
  int lane   = threadIdx.x & 31;
  int wave   = (blockIdx.x * blockDim.x + threadIdx.x) >> 5;
  int nwaves = (gridDim.x * blockDim.x) >> 5;
  float a0 = att[lane], a1 = att[32 + lane];
  for (int e = wave; e < ne; e += nwaves) {
    if (e + nwaves < ne) __builtin_prefetch(edges + 2 * (e + nwaves), 0, 0);
    int src = edges[2 * e];
    int dst = edges[2 * e + 1];
    float fs = srcfeat[(size_t)src * 32 + lane];
    float nd = nemb[(size_t)dst * 32 + lane];
    float s  = wave_sum32(fs * a0 + nd * a1);
    s = (s > 0.f) ? s : 0.1f * expm1f(s);   // ELU(0.1)
    s = expf(s);
    atomicAdd(&msg[(size_t)src * 32 + lane], s * nd);
    if (lane == 0) atomicAdd(&rowsum[src], s);
  }
}

__global__ void finalize_kernel(float* __restrict__ msg,
                                const float* __restrict__ rowsum, int n) {
  int i = blockIdx.x * blockDim.x + threadIdx.x;
  if (i < n) {
    float rs = rowsum[i >> 5];
    msg[i] /= (rs == 0.f ? 1.f : rs);
  }
}

// ---------------- update MLP: [x|m0..m3](160) -> 64 (leaky) -> 32 ----------------
__global__ void __launch_bounds__(128)
update_mlp_kernel(const float* __restrict__ feat_a, const float* __restrict__ feat_b,
                  const float* __restrict__ msg_base, const float* __restrict__ W1,
                  const float* __restrict__ b1, const float* __restrict__ alphap,
                  const float* __restrict__ W2, const float* __restrict__ b2,
                  float* __restrict__ emb) {
  __shared__ __align__(16) _Float16 W1t[64 * 160];
  __shared__ __align__(16) _Float16 W2t[32 * 64];
  __shared__ __align__(16) _Float16 xt[4][16 * 160];
  __shared__ __align__(16) _Float16 ht[4][16 * 64];
  __shared__ float b1s[64], b2s[32];

  int tid = threadIdx.x;
  for (int i = tid; i < 160 * 64; i += blockDim.x) {
    int k = i >> 6, j = i & 63;
    W1t[j * 160 + k] = (_Float16)W1[i];
  }
  for (int i = tid; i < 64 * 32; i += blockDim.x) {
    int k = i >> 5, j = i & 31;
    W2t[j * 64 + k] = (_Float16)W2[i];
  }
  if (tid < 64) b1s[tid] = b1[tid];
  if (tid < 32) b2s[tid] = b2[tid];
  __syncthreads();
  float alpha = alphap[0];

  int lane = tid & 31;
  int wv   = tid >> 5;
  int tile = blockIdx.x * 4 + wv;          // 3125 blocks * 4 waves = 12500, exact
  int side = (tile >= 6250) ? 1 : 0;
  int row0 = (tile - side * 6250) << 4;
  const float* feat = side ? feat_b : feat_a;
  const float* msgs = msg_base + (size_t)side * 4 * (size_t)NA_ * 32;

  // stage x tile (16 x 160) as f16
  _Float16* x = xt[wv];
  for (int r = 0; r < 16; ++r) {
    size_t grow = (size_t)(row0 + r);
    for (int kk = lane; kk < 160; kk += 32) {
      int ch = kk >> 5, d = kk & 31;
      float v = (ch == 0) ? feat[grow * 32 + d]
                          : msgs[(size_t)(ch - 1) * NA_ * 32 + grow * 32 + d];
      x[r * 160 + kk] = (_Float16)v;
    }
  }
  __syncthreads();

  // layer 1: 4 col tiles x 5 K-chunks, then leaky ReLU -> h tile (f16)
  _Float16* h = ht[wv];
#pragma unroll
  for (int t4 = 0; t4 < 4; ++t4) {
    int col = t4 * 16 + (lane & 15);
    v8f c;
    float bv = b1s[col];
#pragma unroll
    for (int j = 0; j < 8; ++j) c[j] = bv;
#pragma unroll
    for (int ch = 0; ch < 5; ++ch) {
      v16h a = load_a_tile(x, 0, 160, ch * 32, lane);
      v16h b = load_b_tile(W1t, t4 * 16, 160, ch * 32, lane);
      c = wmma_f16(a, b, c);
    }
    int rb = (lane >> 4) << 3;
#pragma unroll
    for (int j = 0; j < 8; ++j) {
      float v = c[j];
      v = (v > 0.f) ? v : alpha * v;
      h[(rb + j) * 64 + col] = (_Float16)v;
    }
  }
  __syncthreads();

  // layer 2: 2 col tiles x 2 K-chunks -> emb
  size_t outbase = (size_t)side * NA_ + (size_t)row0;
#pragma unroll
  for (int t2 = 0; t2 < 2; ++t2) {
    int col = t2 * 16 + (lane & 15);
    v8f c;
    float bv = b2s[col];
#pragma unroll
    for (int j = 0; j < 8; ++j) c[j] = bv;
#pragma unroll
    for (int ch = 0; ch < 2; ++ch) {
      v16h a = load_a_tile(h, 0, 64, ch * 32, lane);
      v16h b = load_b_tile(W2t, t2 * 16, 64, ch * 32, lane);
      c = wmma_f16(a, b, c);
    }
    int rb = (lane >> 4) << 3;
#pragma unroll
    for (int j = 0; j < 8; ++j)
      emb[(outbase + rb + j) * 32 + col] = c[j];
  }
}

// ---------------- BPR loss + L2 reg (wave per sample) ----------------
__global__ void __launch_bounds__(256)
loss_kernel(const float* __restrict__ emb, const int* __restrict__ u,
            const int* __restrict__ v, const int* __restrict__ nidx,
            const float* __restrict__ w, float* __restrict__ out) {
  int lane = threadIdx.x & 31;
  int b = (blockIdx.x * blockDim.x + threadIdx.x) >> 5;
  if (b >= B_) return;
  float ue  = emb[(size_t)u[b] * 32 + lane];
  float ve  = emb[(size_t)v[b] * 32 + lane];
  float pos = wave_sum32(ue * ve);
  float reg = ue * ue + ve * ve;
  float wb  = w[b];
  float sgn = (wb > 0.f) ? 1.f : ((wb < 0.f) ? -1.f : 0.f);
  float acc = 0.f;
  for (int k = 0; k < K_; ++k) {
    float ne  = emb[(size_t)nidx[b * K_ + k] * 32 + lane];
    float neg = wave_sum32(ue * ne);
    reg += ne * ne;
    float xx = sgn * ((float)K_ * pos - neg);
    float ls = fminf(xx, 0.f) - log1pf(expf(-fabsf(xx)));  // log_sigmoid
    acc -= ls;
  }
  float regsum = wave_sum32(reg);
  if (lane == 0) atomicAdd(out, acc + REG_ * regsum);
}

extern "C" void kernel_launch(void* const* d_in, const int* in_sizes, int n_in,
                              void* d_out, int out_size, void* d_ws, size_t ws_size,
                              hipStream_t stream) {
  (void)in_sizes; (void)n_in; (void)ws_size;
  const float* feat_a = (const float*)d_in[0];
  const float* feat_b = (const float*)d_in[1];
  const float *W[8], *bvec[8], *att[8];
  for (int r = 0; r < 8; ++r) {
    W[r]    = (const float*)d_in[2 + 3 * r];
    bvec[r] = (const float*)d_in[3 + 3 * r];
    att[r]  = (const float*)d_in[4 + 3 * r];
  }
  const float* W1    = (const float*)d_in[26];
  const float* b1    = (const float*)d_in[27];
  const float* alpha = (const float*)d_in[28];
  const float* W2    = (const float*)d_in[29];
  const float* b2    = (const float*)d_in[30];
  const int* edges[8];
  for (int r = 0; r < 8; ++r) edges[r] = (const int*)d_in[31 + r];
  const int*   u    = (const int*)d_in[39];
  const int*   v    = (const int*)d_in[40];
  const int*   nidx = (const int*)d_in[41];
  const float* w    = (const float*)d_in[42];
  float* out = (float*)d_out;

  // Workspace layout (floats): new_emb | msg[8] | emb | rowsum | f16 feats
  const size_t NF = (size_t)NA_ * 32;   // 3.2M
  float* ws      = (float*)d_ws;
  float* nemb    = ws;                  // NF
  float* msg     = ws + NF;             // 8*NF
  float* emb     = ws + 9 * NF;         // 2*NF
  float* rowsum  = ws + 11 * NF;        // NA_
  _Float16* f16a = (_Float16*)(ws + 11 * NF + (size_t)NA_);
  _Float16* f16b = f16a + NF;

  hipMemsetAsync(d_out, 0, sizeof(float) * (size_t)out_size, stream);

  int n = (int)NF;
  cvt_f16_kernel<<<(n + 255) / 256, 256, 0, stream>>>(feat_a, f16a, n);
  cvt_f16_kernel<<<(n + 255) / 256, 256, 0, stream>>>(feat_b, f16b, n);

  // NAMES order: ab_pos, ab_neg, aa_pos, aa_neg, ba_pos, ba_neg, bb_pos, bb_neg
  const _Float16* nbr16[8] = { f16b, f16b, f16a, f16a, f16a, f16a, f16b, f16b };
  const float*    srcf[8]  = { feat_a, feat_a, feat_a, feat_a,
                               feat_b, feat_b, feat_b, feat_b };

  for (int r = 0; r < 8; ++r) {
    float* msg_r = msg + (size_t)r * NF;
    transform_kernel<<<196, 256, 0, stream>>>(nbr16[r], W[r], bvec[r], nemb, NA_);
    hipMemsetAsync(msg_r, 0, NF * sizeof(float), stream);
    hipMemsetAsync(rowsum, 0, (size_t)NA_ * sizeof(float), stream);
    edge_agg_kernel<<<1024, 256, 0, stream>>>(edges[r], srcf[r], nemb, att[r],
                                              msg_r, rowsum, E_);
    finalize_kernel<<<((int)NF + 255) / 256, 256, 0, stream>>>(msg_r, rowsum, (int)NF);
  }

  update_mlp_kernel<<<3125, 128, 0, stream>>>(feat_a, feat_b, msg, W1, b1, alpha,
                                              W2, b2, emb);
  loss_kernel<<<(B_ * 32 + 255) / 256, 256, 0, stream>>>(emb, u, v, nidx, w, out);
}
